// Model_22548578304599
// MI455X (gfx1250) — matrix operand order repaired and checked by reading, not compiled, not run on hardware
//
#include <hip/hip_runtime.h>
#include <math.h>

// ---------------------------------------------------------------------------
// Batched pairwise L2 distance via bf16x2-split WMMA GEMM (gfx1250 / CDNA5).
//   out[b,a,v] = sqrt(max(asq[a] + vsq[v] - 2 * dot(audio[b,a], visual[b,v]), 0))
// cross term: v_wmma_f32_16x16x32_bf16 on the hi/lo bf16 split of the fp32
// inputs (3 WMMAs per K-chunk) -> near-fp32 accuracy at bf16 matrix rate.
// Global->LDS staging uses the CDNA5 async-to-LDS pipe (ASYNCcnt), double
// buffered with lookahead-1 prefetch.
// ---------------------------------------------------------------------------

typedef __attribute__((ext_vector_type(16))) __bf16         v16bf;
typedef __attribute__((ext_vector_type(8)))  float          v8f;
typedef __attribute__((ext_vector_type(4)))  unsigned short ushort4v;
typedef __attribute__((ext_vector_type(4)))  int            v4i;

#define B_  16
#define A_  1024
#define V_  1024
#define D_  512

#define BM  128            // block tile M (audio rows)
#define BN  128            // block tile N (visual rows)
#define KT  32             // K chunk (one bf16 WMMA K)
#define NCHUNK (D_ / KT)
#define LSTR 40            // bf16 LDS row stride in ushorts (80B: aligned + conflict-free)

#define STAGE_BYTES (2 * BM * KT * 4)   // fp32 staging per buffer: A tile + B tile = 32KB

struct Pack32 { uint4 a, b; };

__device__ __forceinline__ v16bf make_frag(uint4 a, uint4 b) {
    Pack32 p{a, b};
    return __builtin_bit_cast(v16bf, p);
}

// ---- CDNA5 async Global->LDS (ASYNCcnt) ----------------------------------
#if defined(__has_builtin)
#if __has_builtin(__builtin_amdgcn_global_load_async_to_lds_b128)
#define HAVE_ASYNC_BUILTIN 1
#endif
#if __has_builtin(__builtin_amdgcn_s_wait_asynccnt)
#define HAVE_WAITASYNC_BUILTIN 1
#endif
#endif

__device__ __forceinline__ void async_b128(const float* gsrc, char* ldst) {
#ifdef HAVE_ASYNC_BUILTIN
    // Builtin signature (from clang diagnostic): param0 = AS1 int4*, non-const.
    __builtin_amdgcn_global_load_async_to_lds_b128(
        (__attribute__((address_space(1))) v4i*)(uintptr_t)gsrc,
        (__attribute__((address_space(3))) v4i*)(unsigned)(uintptr_t)ldst,
        0, 0);
#else
    asm volatile("global_load_async_to_lds_b128 %0, %1, off"
                 :: "v"((unsigned)(uintptr_t)ldst),
                    "v"((unsigned long long)(uintptr_t)gsrc)
                 : "memory");
#endif
}

#ifdef HAVE_WAITASYNC_BUILTIN
#define WAIT_ASYNC(n) __builtin_amdgcn_s_wait_asynccnt(n)
#else
#define WAIT_ASYNC(n) asm volatile("s_wait_asynccnt %0" :: "n"(n) : "memory")
#endif

// ---------------------------------------------------------------------------
// Kernel 1: fp32 row norms -> d_ws. ws[0..16384)=|audio row|^2,
// ws[16384..32768)=|visual row|^2. One wave per row; also warms L2 (192MB
// holds both 32MB inputs entirely).
// ---------------------------------------------------------------------------
__global__ __launch_bounds__(256) void norms_kernel(
    const float* __restrict__ audio, const float* __restrict__ visual,
    float* __restrict__ ws) {
    const int wave = (blockIdx.x * blockDim.x + threadIdx.x) >> 5;
    const int lane = threadIdx.x & 31;
    const float* src = (wave < B_ * A_)
        ? audio  + (size_t)wave * D_
        : visual + (size_t)(wave - B_ * A_) * D_;
    float s = 0.0f;
    #pragma unroll
    for (int j = 0; j < D_ / 32; ++j) {
        float x = src[lane + j * 32];
        s = fmaf(x, x, s);
    }
    #pragma unroll
    for (int m = 16; m >= 1; m >>= 1) s += __shfl_xor(s, m, 32);
    if (lane == 0) ws[wave] = s;
}

// ---------------------------------------------------------------------------
// Kernel 2: tiled GEMM + distance epilogue.
//   static LDS : bf16 regions A-hi|A-lo|B-hi|B-lo (40KB)
//   dynamic LDS: fp32 staging, double buffered (2 x 32KB)
// ---------------------------------------------------------------------------
__global__ __launch_bounds__(256) void l2dist_kernel(
    const float* __restrict__ audio, const float* __restrict__ visual,
    const float* __restrict__ norms, float* __restrict__ out) {

    __shared__ unsigned short bfsm[4 * BM * LSTR];      // 40KB
    extern __shared__ __align__(16) char stage[];       // 64KB (set at launch)

    constexpr int AHI = 0;
    constexpr int ALO = BM * LSTR;
    constexpr int BHI = 2 * BM * LSTR;
    constexpr int BLO = 3 * BM * LSTR;

    const int b    = blockIdx.z;
    const int m0   = blockIdx.x * BM;
    const int n0   = blockIdx.y * BN;
    const int tid  = threadIdx.x;
    const int wave = tid >> 5;
    const int lane = tid & 31;
    const int lh   = lane >> 4;   // half-wave select
    const int lr   = lane & 15;
    const int wm   = wave & 3;    // M wave coord: rows wm*32 .. +31
    const int wn   = wave >> 2;   // N wave coord: rows wn*64 .. +63

    const float* aBase = audio  + ((size_t)b * A_ + m0) * D_;
    const float* vBase = visual + ((size_t)b * V_ + n0) * D_;

    v8f acc[2][4] = {};

    // Per-thread slice: 4 x float4 per matrix per chunk (8 async b128 / chunk).
    // idx = tid + i*256 -> row = idx>>3 (0..127), c4 = idx&7.
    auto issue_chunk = [&](int kc) {
        char* dstA = stage + (size_t)(kc & 1) * STAGE_BYTES;
        char* dstB = dstA + BM * KT * 4;
        #pragma unroll
        for (int i = 0; i < 4; ++i) {
            const int idx = tid + i * 256;
            const int row = idx >> 3;
            const int c4  = idx & 7;
            const size_t goff = (size_t)row * D_ + kc * KT + c4 * 4;
            const int    loff = (row * KT + c4 * 4) * 4;
            async_b128(aBase + goff, dstA + loff);
            async_b128(vBase + goff, dstB + loff);
        }
    };

    auto split_store = [&](const float4& v, int base, int row, int c4) {
        float f[4] = {v.x, v.y, v.z, v.w};
        ushort4v hi, lo;
        #pragma unroll
        for (int j = 0; j < 4; ++j) {
            __bf16 h = (__bf16)f[j];
            __bf16 l = (__bf16)(f[j] - (float)h);
            hi[j] = __builtin_bit_cast(unsigned short, h);
            lo[j] = __builtin_bit_cast(unsigned short, l);
        }
        const int o = row * LSTR + c4 * 4;                   // 8B aligned
        *(ushort4v*)&bfsm[base + o]             = hi;
        *(ushort4v*)&bfsm[base + BM * LSTR + o] = lo;        // lo region follows hi
    };

    // Convert own staged slice (fp32) -> bf16 hi/lo regions.
    auto convert_chunk = [&](int kc) {
        const float* stA = (const float*)(stage + (size_t)(kc & 1) * STAGE_BYTES);
        const float* stB = stA + BM * KT;
        #pragma unroll
        for (int i = 0; i < 4; ++i) {
            const int idx = tid + i * 256;
            const int row = idx >> 3;
            const int c4  = idx & 7;
            const int off = row * KT + c4 * 4;
            float4 a = *(const float4*)(stA + off);
            float4 v = *(const float4*)(stB + off);
            split_store(a, AHI, row, c4);
            split_store(v, BHI, row, c4);
        }
    };

    auto compute_chunk = [&]() {
        // A fragments (16x32 bf16, ISA 7.12.2): K bases {lh*8, 16+lh*8}
        v16bf fahi[2], falo[2];
        #pragma unroll
        for (int mt = 0; mt < 2; ++mt) {
            const int row = wm * 32 + mt * 16 + lr;
            const int off = row * LSTR + lh * 8;
            fahi[mt] = make_frag(*(const uint4*)&bfsm[AHI + off],
                                 *(const uint4*)&bfsm[AHI + off + 16]);
            falo[mt] = make_frag(*(const uint4*)&bfsm[ALO + off],
                                 *(const uint4*)&bfsm[ALO + off + 16]);
        }
        #pragma unroll
        for (int nt = 0; nt < 4; ++nt) {
            // B fragments (32x16 bf16): lane = N, K bases {lh*8, 16+lh*8}
            const int row = wn * 64 + nt * 16 + lr;
            const int off = row * LSTR + lh * 8;
            v16bf fbhi = make_frag(*(const uint4*)&bfsm[BHI + off],
                                   *(const uint4*)&bfsm[BHI + off + 16]);
            v16bf fblo = make_frag(*(const uint4*)&bfsm[BLO + off],
                                   *(const uint4*)&bfsm[BLO + off + 16]);
            #pragma unroll
            for (int mt = 0; mt < 2; ++mt) {
                acc[mt][nt] = __builtin_amdgcn_wmma_f32_16x16x32_bf16(
                    false, fahi[mt], false, fbhi, (short)0, acc[mt][nt], false, false);
                acc[mt][nt] = __builtin_amdgcn_wmma_f32_16x16x32_bf16(
                    false, fahi[mt], false, fblo, (short)0, acc[mt][nt], false, false);
                acc[mt][nt] = __builtin_amdgcn_wmma_f32_16x16x32_bf16(
                    false, falo[mt], false, fbhi, (short)0, acc[mt][nt], false, false);
            }
        }
    };

    // ---- pipelined K loop: async chunk k+1 in flight while converting /
    //      computing chunk k. Each thread converts exactly the slice it
    //      loaded, so only its own ASYNCcnt wait + the WG barrier are needed.
    issue_chunk(0);
    #pragma unroll 1
    for (int kc = 0; kc < NCHUNK; ++kc) {
        if (kc + 1 < NCHUNK) {
            issue_chunk(kc + 1);     // buffer (kc+1)&1 was consumed at kc-1
            WAIT_ASYNC(8);           // in-order completion: chunk kc arrived
        } else {
            WAIT_ASYNC(0);
        }
        convert_chunk(kc);           // bf16 region free: compute kc-1 done
        __syncthreads();             // bf16 visible to all waves
        compute_chunk();
        __syncthreads();             // all reads done; region reusable
    }

    // ---- epilogue: dist = sqrt(max(asq + vsq - 2c, 0))
    // C layout (ISA 7.12.2): VGPR r -> M = base + r + 8*lh, N = nbase + lr
    const float* asq = norms + (size_t)b * A_;
    const float* vsq = norms + (size_t)B_ * A_ + (size_t)b * V_;
    #pragma unroll
    for (int mt = 0; mt < 2; ++mt) {
        #pragma unroll
        for (int nt = 0; nt < 4; ++nt) {
            const int n   = n0 + wn * 64 + nt * 16 + lr;
            const float vs = vsq[n];
            #pragma unroll
            for (int r = 0; r < 8; ++r) {
                const int m = m0 + wm * 32 + mt * 16 + lh * 8 + r;
                float d2 = asq[m] + vs - 2.0f * acc[mt][nt][r];
                d2 = fmaxf(d2, 0.0f);
                out[((size_t)b * A_ + m) * V_ + n] = sqrtf(d2);
            }
        }
    }
}

// ---------------------------------------------------------------------------
extern "C" void kernel_launch(void* const* d_in, const int* in_sizes, int n_in,
                              void* d_out, int out_size, void* d_ws, size_t ws_size,
                              hipStream_t stream) {
    (void)in_sizes; (void)n_in; (void)out_size; (void)ws_size;
    const float* audio  = (const float*)d_in[0];
    const float* visual = (const float*)d_in[1];
    float*       out    = (float*)d_out;
    float*       ws     = (float*)d_ws;   // 32768 floats = 128KB of norms

    norms_kernel<<<(B_ * (A_ + V_)) / 8, 256, 0, stream>>>(audio, visual, ws);

    dim3 grid(A_ / BM, V_ / BN, B_);
    l2dist_kernel<<<grid, 256, 2 * STAGE_BYTES, stream>>>(audio, visual, ws, out);
}
